// SelfAttention_71871982731568
// MI455X (gfx1250) — matrix operand order repaired and checked by reading, not compiled, not run on hardware
//
#include <hip/hip_runtime.h>

// ---------------------------------------------------------------------------
// MI455X (gfx1250) causal self-attention, fp32 I/O, bf16 WMMA compute path.
// All matmuls (QKV proj, Q*K^T, P*V, O proj) use v_wmma_f32_16x16x32_bf16.
// K tiles staged into LDS with GLOBAL_LOAD_ASYNC_TO_LDS (ASYNCcnt).
// ---------------------------------------------------------------------------

typedef __attribute__((ext_vector_type(16))) __bf16 v16bf;
typedef __attribute__((ext_vector_type(8)))  __bf16 v8bf;
typedef __attribute__((ext_vector_type(2)))  __bf16 v2bf;
typedef __attribute__((ext_vector_type(8)))  float  v8f;
typedef int v4i_vs __attribute__((vector_size(16)));            // gcc-style int4
typedef v4i_vs __attribute__((address_space(1)))* gptr_b128;    // global int4*
typedef v4i_vs __attribute__((address_space(3)))* lptr_b128;    // LDS int4*

#if __has_builtin(__builtin_amdgcn_global_load_async_to_lds_b128)
#define USE_ASYNC_LDS 1
#else
#define USE_ASYNC_LDS 0
#endif

#define DEV __device__ __forceinline__

static constexpr int T  = 4096;
static constexpr int DM = 1024;
static constexpr int NH = 16;
static constexpr int DH = 64;

DEV v16bf cat8(v8bf lo, v8bf hi) {
  return __builtin_shufflevector(lo, hi, 0,1,2,3,4,5,6,7,8,9,10,11,12,13,14,15);
}

// A-matrix 16x32 bf16 tile from row-major [rows x ld]:
// lane m(0..15) holds row r0+m with K={koff..koff+7, koff+16..koff+23},
// koff = (lane>=16)*8  (ISA 7.12.2 16-bit A layout).
DEV v16bf load_a(const __bf16* base, int ld, int r0, int c0, int lane) {
  int l16 = lane & 15, hf = lane >> 4;
  const __bf16* p = base + (size_t)(r0 + l16) * ld + c0 + hf * 8;
  v8bf lo = *(const v8bf*)p;
  v8bf hi = *(const v8bf*)(p + 16);
  return cat8(lo, hi);
}

// B-matrix 32x16 bf16 tile where B[k][n] = Wrow[n0+n][c0+k], W row-major [N x ld]:
// lane n(0..15) holds column n0+n with k = {kb..kb+7, kb+16..kb+23}, kb = c0 + (lane>=16)*8.
DEV v16bf load_b(const __bf16* base, int ld, int n0, int c0, int lane) {
  int l16 = lane & 15, hf = lane >> 4;
  const __bf16* p = base + (size_t)(n0 + l16) * ld + c0 + hf * 8;
  v8bf lo = *(const v8bf*)p;
  v8bf hi = *(const v8bf*)(p + 16);
  return cat8(lo, hi);
}

DEV v8f wmma_bf16(v16bf a, v16bf b, v8f c) {
  return __builtin_amdgcn_wmma_f32_16x16x32_bf16(false, a, false, b, (short)0, c,
                                                 false, false);
}

// ---------------------------------------------------------------------------
__global__ void cvt_f32_bf16(const float* __restrict__ s, __bf16* __restrict__ d,
                             int n) {
  int i = blockIdx.x * 256 + threadIdx.x;
  if (i < n) d[i] = (__bf16)s[i];
}

// ---------------------------------------------------------------------------
// Fused QKV projection: y = x @ W^T + b, written to per-head (h,t,d) bf16.
// Block = 256 threads = 8 waves; block tile 128(M) x 256(N); wave tile 64x64.
// Grid: (T/128, 3*DM/256).  Q is pre-scaled by 1/sqrt(DH)=0.125.
__global__ __launch_bounds__(256) void qkv_gemm(
    const __bf16* __restrict__ xb,
    const __bf16* __restrict__ wq, const __bf16* __restrict__ wk,
    const __bf16* __restrict__ wv,
    const float* __restrict__ bq, const float* __restrict__ bk,
    const float* __restrict__ bv,
    __bf16* __restrict__ Qb, __bf16* __restrict__ Kb, __bf16* __restrict__ Vb) {
  const int wave = threadIdx.x >> 5, lane = threadIdx.x & 31;
  const int l16 = lane & 15, hf = lane >> 4;
  const int wm = wave & 1, wn = wave >> 1;
  const int r0  = blockIdx.x * 128 + wm * 64;
  const int n0g = blockIdx.y * 256 + wn * 64;             // 0..3071
  const int mat = n0g >> 10;                              // 0=Q 1=K 2=V
  const int ob  = n0g & (DM - 1);
  const __bf16* W    = (mat == 0) ? wq : (mat == 1) ? wk : wv;
  const float*  bias = (mat == 0) ? bq : (mat == 1) ? bk : bv;
  __bf16*       dst  = (mat == 0) ? Qb : (mat == 1) ? Kb : Vb;
  const float scale = (mat == 0) ? 0.125f : 1.0f;

  v8f acc[4][4] = {};
  for (int k0 = 0; k0 < DM; k0 += 32) {
    v16bf a[4], b[4];
#pragma unroll
    for (int i = 0; i < 4; i++) a[i] = load_a(xb, DM, r0 + i * 16, k0, lane);
#pragma unroll
    for (int i = 0; i < 4; i++) b[i] = load_b(W, DM, ob + i * 16, k0, lane);
#pragma unroll
    for (int mi = 0; mi < 4; mi++)
#pragma unroll
      for (int ni = 0; ni < 4; ni++)
        acc[mi][ni] = wmma_bf16(a[mi], b[ni], acc[mi][ni]);
  }
#pragma unroll
  for (int mi = 0; mi < 4; mi++)
#pragma unroll
    for (int ni = 0; ni < 4; ni++)
#pragma unroll
      for (int j = 0; j < 8; j++) {
        int t = r0 + mi * 16 + j + 8 * hf;                // C layout row
        int o = ob + ni * 16 + l16;                       // C layout col
        float v = (acc[mi][ni][j] + bias[o]) * scale;
        int hh = o >> 6, d = o & 63;
        dst[((size_t)hh * T + t) * DH + d] = (__bf16)v;
      }
}

// ---------------------------------------------------------------------------
// Causal flash attention. Grid: (T/64, NH); block = 128 threads = 4 waves.
// Wave w owns q rows [q0+16w, q0+16w+16). Uniform key-block count per block.
__global__ __launch_bounds__(128) void attn(
    const __bf16* __restrict__ Qb, const __bf16* __restrict__ Kb,
    const __bf16* __restrict__ Vb, __bf16* __restrict__ Yb) {
  __shared__ __bf16 Kt[32][72];        // K block, row-major (key, d), padded
  __shared__ __bf16 Vt[64][40];        // V block transposed (d, key), padded
  __shared__ __bf16 Pt[4][16][40];     // per-wave P staging (q, key), padded

  const int h  = blockIdx.y;
  const int q0 = blockIdx.x * 64;
  const int wave = threadIdx.x >> 5, lane = threadIdx.x & 31;
  const int l16 = lane & 15, hf = lane >> 4;
  const int qw = q0 + wave * 16;

  const __bf16* Qh = Qb + (size_t)h * T * DH;
  const __bf16* Kh = Kb + (size_t)h * T * DH;
  const __bf16* Vh = Vb + (size_t)h * T * DH;

  const v16bf qa0 = load_a(Qh, DH, qw, 0, lane);   // Q rows, d 0..31
  const v16bf qa1 = load_a(Qh, DH, qw, 32, lane);  // Q rows, d 32..63

  v8f acc[4] = {};
  float mrun[8], lrun[8];
#pragma unroll
  for (int j = 0; j < 8; j++) { mrun[j] = -1e30f; lrun[j] = 0.0f; }

  const int nkb = (q0 >> 5) + 2;                   // keys 0 .. q0+63
  for (int kb = 0; kb < nkb; kb++) {
    const int j0 = kb * 32;
    __syncthreads();
    {
      // --- K tile: async DMA global -> LDS (no VGPR round trip) ---
      int kl = threadIdx.x >> 2, seg = threadIdx.x & 3;   // 32 rows x 4 segs
      const __bf16* ksrc = Kh + (size_t)(j0 + kl) * DH + seg * 16;
      __bf16* kdst = &Kt[kl][seg * 16];
#if USE_ASYNC_LDS
      __builtin_amdgcn_global_load_async_to_lds_b128(
          (gptr_b128)ksrc, (lptr_b128)kdst, 0, 0);
      __builtin_amdgcn_global_load_async_to_lds_b128(
          (gptr_b128)(ksrc + 8), (lptr_b128)(kdst + 8), 0, 0);
#else
      const v8bf* ks = (const v8bf*)ksrc;
      *(v8bf*)kdst       = ks[0];
      *(v8bf*)(kdst + 8) = ks[1];
#endif
      // --- V tile: transpose while staging (overlaps with async K DMA) ---
      // thread handles 2 adjacent keys x 8 d -> packed 2xbf16 LDS stores
      int kp = (threadIdx.x >> 3) << 1;                   // even key 0..30
      int d0 = (threadIdx.x & 7) * 8;
      v8bf va = *(const v8bf*)(Vh + (size_t)(j0 + kp)     * DH + d0);
      v8bf vb = *(const v8bf*)(Vh + (size_t)(j0 + kp + 1) * DH + d0);
#pragma unroll
      for (int e = 0; e < 8; e++) {
        v2bf pv = {va[e], vb[e]};
        *(v2bf*)&Vt[d0 + e][kp] = pv;
      }
      // prefetch next key block (K/V are L2-resident; keep WGP latency hidden)
      if (kb + 1 < nkb) {
        __builtin_prefetch(Kh + (size_t)(j0 + 32 + kl) * DH + seg * 16, 0, 1);
        __builtin_prefetch(Vh + (size_t)(j0 + 32 + kl) * DH + seg * 16, 0, 1);
      }
#if USE_ASYNC_LDS
      asm volatile("s_wait_asynccnt 0" ::: "memory");
#endif
    }
    __syncthreads();

    // S = Q K^T for two 16-key chunks
    v8f s[2];
#pragma unroll
    for (int c = 0; c < 2; c++) {
      v16bf b0 = load_b(&Kt[0][0], 72, c * 16, 0,  lane);
      v16bf b1 = load_b(&Kt[0][0], 72, c * 16, 32, lane);
      v8f z = {};
      z = wmma_bf16(qa0, b0, z);
      s[c] = wmma_bf16(qa1, b1, z);
    }

    // causal mask + block row max
    float mblk[8];
#pragma unroll
    for (int j = 0; j < 8; j++) mblk[j] = -1e30f;
#pragma unroll
    for (int c = 0; c < 2; c++)
#pragma unroll
      for (int j = 0; j < 8; j++) {
        int key = j0 + c * 16 + l16;
        int qr  = qw + j + 8 * hf;
        if (key > qr) s[c][j] = -1e30f;
        mblk[j] = fmaxf(mblk[j], s[c][j]);
      }
#pragma unroll
    for (int j = 0; j < 8; j++)
#pragma unroll
      for (int off = 1; off < 16; off <<= 1)
        mblk[j] = fmaxf(mblk[j], __shfl_xor(mblk[j], off, 16));

    float alpha[8], psum[8];
#pragma unroll
    for (int j = 0; j < 8; j++) {
      float mn = fmaxf(mrun[j], mblk[j]);
      alpha[j] = __expf(mrun[j] - mn);
      mrun[j]  = mn;
      psum[j]  = 0.0f;
    }
    // P = exp(S - m), row sums
#pragma unroll
    for (int c = 0; c < 2; c++)
#pragma unroll
      for (int j = 0; j < 8; j++) {
        float p = __expf(s[c][j] - mrun[j]);
        s[c][j] = p;
        psum[j] += p;
      }
#pragma unroll
    for (int j = 0; j < 8; j++) {
#pragma unroll
      for (int off = 1; off < 16; off <<= 1)
        psum[j] += __shfl_xor(psum[j], off, 16);
      lrun[j] = lrun[j] * alpha[j] + psum[j];
    }
    // rescale O accumulators
#pragma unroll
    for (int t = 0; t < 4; t++)
#pragma unroll
      for (int j = 0; j < 8; j++) acc[t][j] *= alpha[j];

    // P: C layout -> LDS (per-wave region) -> A layout
#pragma unroll
    for (int c = 0; c < 2; c++)
#pragma unroll
      for (int j = 0; j < 8; j++)
        Pt[wave][j + 8 * hf][c * 16 + l16] = (__bf16)s[c][j];
    asm volatile("s_wait_dscnt 0" ::: "memory");   // same-wave LDS RAW

    v16bf pa = load_a(&Pt[wave][0][0], 40, 0, 0, lane);
#pragma unroll
    for (int t = 0; t < 4; t++) {
      v16bf vbm = load_b(&Vt[0][0], 40, t * 16, 0, lane);
      acc[t] = wmma_bf16(pa, vbm, acc[t]);
    }
  }

  // normalize and write attention output (t, h*64+d) as bf16
#pragma unroll
  for (int j = 0; j < 8; j++) lrun[j] = 1.0f / lrun[j];
#pragma unroll
  for (int t = 0; t < 4; t++)
#pragma unroll
    for (int j = 0; j < 8; j++) {
      int qr = qw + j + 8 * hf;
      int d  = t * 16 + l16;
      Yb[(size_t)qr * DM + h * DH + d] = (__bf16)(acc[t][j] * lrun[j]);
    }
}

// ---------------------------------------------------------------------------
// Output projection: out = Y @ Wo^T + bo (f32 out). Grid: (T/128, DM/256).
__global__ __launch_bounds__(256) void oproj_gemm(
    const __bf16* __restrict__ Yb, const __bf16* __restrict__ Wo,
    const float* __restrict__ bo, float* __restrict__ out) {
  const int wave = threadIdx.x >> 5, lane = threadIdx.x & 31;
  const int l16 = lane & 15, hf = lane >> 4;
  const int wm = wave & 1, wn = wave >> 1;
  const int r0 = blockIdx.x * 128 + wm * 64;
  const int n0 = blockIdx.y * 256 + wn * 64;

  v8f acc[4][4] = {};
  for (int k0 = 0; k0 < DM; k0 += 32) {
    v16bf a[4], b[4];
#pragma unroll
    for (int i = 0; i < 4; i++) a[i] = load_a(Yb, DM, r0 + i * 16, k0, lane);
#pragma unroll
    for (int i = 0; i < 4; i++) b[i] = load_b(Wo, DM, n0 + i * 16, k0, lane);
#pragma unroll
    for (int mi = 0; mi < 4; mi++)
#pragma unroll
      for (int ni = 0; ni < 4; ni++)
        acc[mi][ni] = wmma_bf16(a[mi], b[ni], acc[mi][ni]);
  }
#pragma unroll
  for (int mi = 0; mi < 4; mi++)
#pragma unroll
    for (int ni = 0; ni < 4; ni++)
#pragma unroll
      for (int j = 0; j < 8; j++) {
        int t = r0 + mi * 16 + j + 8 * hf;
        int o = n0 + ni * 16 + l16;
        out[(size_t)t * DM + o] = acc[mi][ni][j] + bo[o];
      }
}

// ---------------------------------------------------------------------------
extern "C" void kernel_launch(void* const* d_in, const int* in_sizes, int n_in,
                              void* d_out, int out_size, void* d_ws,
                              size_t ws_size, hipStream_t stream) {
  const float* x  = (const float*)d_in[0];
  const float* Wq = (const float*)d_in[1];
  const float* bq = (const float*)d_in[2];
  const float* Wk = (const float*)d_in[3];
  const float* bk = (const float*)d_in[4];
  const float* Wv = (const float*)d_in[5];
  const float* bv = (const float*)d_in[6];
  const float* Wo = (const float*)d_in[7];
  const float* bo = (const float*)d_in[8];

  char* ws = (char*)d_ws;
  const size_t MB = 1024u * 1024u;
  __bf16* xb  = (__bf16*)(ws + 0);          //  8 MB (T x DM)
  __bf16* wqb = (__bf16*)(ws + 8 * MB);     //  2 MB each
  __bf16* wkb = (__bf16*)(ws + 10 * MB);
  __bf16* wvb = (__bf16*)(ws + 12 * MB);
  __bf16* wob = (__bf16*)(ws + 14 * MB);
  __bf16* Qb  = (__bf16*)(ws + 16 * MB);    //  8 MB (H,T,DH), pre-scaled
  __bf16* Kb  = (__bf16*)(ws + 24 * MB);
  __bf16* Vb  = (__bf16*)(ws + 32 * MB);
  __bf16* Yb  = (__bf16*)(ws + 40 * MB);    //  8 MB attention output
  // total workspace: 48 MB

  const int nx = T * DM, nw = DM * DM;
  cvt_f32_bf16<<<(nx + 255) / 256, 256, 0, stream>>>(x, xb, nx);
  cvt_f32_bf16<<<(nw + 255) / 256, 256, 0, stream>>>(Wq, wqb, nw);
  cvt_f32_bf16<<<(nw + 255) / 256, 256, 0, stream>>>(Wk, wkb, nw);
  cvt_f32_bf16<<<(nw + 255) / 256, 256, 0, stream>>>(Wv, wvb, nw);
  cvt_f32_bf16<<<(nw + 255) / 256, 256, 0, stream>>>(Wo, wob, nw);

  qkv_gemm<<<dim3(T / 128, (3 * DM) / 256), 256, 0, stream>>>(
      xb, wqb, wkb, wvb, bq, bk, bv, Qb, Kb, Vb);

  attn<<<dim3(T / 64, NH), 128, 0, stream>>>(Qb, Kb, Vb, Yb);

  oproj_gemm<<<dim3(T / 128, DM / 256), 256, 0, stream>>>(Yb, wob, bo,
                                                          (float*)d_out);
}
